// PF_LSTM_54614804136127
// MI455X (gfx1250) — matrix operand order repaired and checked by reading, not compiled, not run on hardware
//
#include <hip/hip_runtime.h>
#include <math.h>

#define LS 2
#define BS 16
#define NHID 256
#define NPART 512
#define TSTEPS 16
#define DIN 64
#define MID 256

typedef __attribute__((ext_vector_type(16))) __bf16 v16bf;
typedef __attribute__((ext_vector_type(8)))  float  v8f;

union BF16x16 { v16bf v; uint4 q[2]; };

__device__ __forceinline__ unsigned short f2bf(float f) {
  unsigned int u = __float_as_uint(f);
  unsigned int r = (u + 0x7FFFu + ((u >> 16) & 1u)) >> 16;  // RNE
  return (unsigned short)r;
}
__device__ __forceinline__ float sigm(float x)   { return 1.0f / (1.0f + __expf(-x)); }
__device__ __forceinline__ float tanh_f(float x) { return 2.0f / (1.0f + __expf(-2.0f * x)) - 1.0f; }

// A-fragment (16x32 bf16, M x K) per CDNA5 ISA layout:
// lane l: m = l&15, g = l>>4; elements 0..7 -> K = k0+8g .. k0+8g+7 ; 8..15 -> K = k0+16+8g ..
__device__ __forceinline__ v16bf load_a_frag(const unsigned short* rowptr) {
  BF16x16 u;
  u.q[0] = *(const uint4*)(rowptr);
  u.q[1] = *(const uint4*)(rowptr + 16);
  return u.v;
}
// B-fragment (32x16 bf16, K x N): lane holds 16 contiguous packed bf16 (pre-packed layout)
__device__ __forceinline__ v16bf load_b_frag(const unsigned short* p) {
  BF16x16 u;
  u.q[0] = *(const uint4*)(p);
  u.q[1] = *(const uint4*)(p + 8);
  return u.v;
}

// ---------------- prep: pack fp32 weights into bf16 B-fragment order ----------------
// layer0: K = 64 (Wih0) + 256 (Whh0) -> 10 ksteps ; layer1: 256 (Wih1) + 256 (Whh1) -> 16 ksteps
// packed index = ((kk*64 + tile)*32 + lane)*16 + j ; col = 16*tile + (lane&15); k = 32*kk + 8*(lane>>4) + (j&7) + 16*(j>>3)
__global__ void pack_weights_kernel(const float* __restrict__ Wih0, const float* __restrict__ Whh0,
                                    const float* __restrict__ Wih1, const float* __restrict__ Whh1,
                                    unsigned short* __restrict__ wpk0, unsigned short* __restrict__ wpk1) {
  int id = blockIdx.x * 256 + threadIdx.x;
  const int N0 = 10 * 64 * 32 * 16;   // 327680
  const int N1 = 16 * 64 * 32 * 16;   // 524288
  if (id < N0) {
    int j = id & 15, lane = (id >> 4) & 31, tl = (id >> 9) & 63, kk = id >> 15;
    int col = 16 * tl + (lane & 15);
    int k = 32 * kk + 8 * (lane >> 4) + (j & 7) + 16 * (j >> 3);
    float v = (k < 64) ? Wih0[col * 64 + k] : Whh0[col * 256 + (k - 64)];
    wpk0[id] = f2bf(v);
  } else {
    int id2 = id - N0;
    if (id2 >= N1) return;
    int j = id2 & 15, lane = (id2 >> 4) & 31, tl = (id2 >> 9) & 63, kk = id2 >> 15;
    int col = 16 * tl + (lane & 15);
    int k = 32 * kk + 8 * (lane >> 4) + (j & 7) + 16 * (j >> 3);
    float v = (k < 256) ? Wih1[col * 256 + k] : Whh1[col * 256 + (k - 256)];
    wpk1[id2] = f2bf(v);
  }
}

// ---------------- main LSTM particle kernel: 1 block = 1 particle, 8 waves ----------------
__global__ __launch_bounds__(256) void lstm_particle_kernel(
    const float* __restrict__ x, const float* __restrict__ particles,
    const float* __restrict__ qp, const float* __restrict__ ep,
    const unsigned short* __restrict__ wpk0, const float* __restrict__ b0,
    const unsigned short* __restrict__ wpk1, const float* __restrict__ b1,
    const float* __restrict__ eps_h, const float* __restrict__ eps_c,
    float* __restrict__ pr) {
  __shared__ alignas(16) unsigned short xbf[TSTEPS][BS][72];  // pitch 72 to kill bank conflicts
  __shared__ alignas(16) unsigned short h0bf[BS][264];        // pitch 264
  __shared__ alignas(16) unsigned short h1bf[BS][264];

  const int p = blockIdx.x;
  const int tid = threadIdx.x;
  const int w = tid >> 5;          // wave id 0..7
  const int lane = tid & 31;
  const int nlo = lane & 15;
  const int g = lane >> 4;

  // x -> LDS bf16 : x is (BS, T, DIN)
  for (int i = tid; i < BS * TSTEPS * DIN; i += 256) {
    int d = i & 63, t = (i >> 6) & 15, b = i >> 10;
    xbf[t][b][d] = f2bf(x[(b * TSTEPS + t) * DIN + d]);
  }
  // h init -> LDS bf16 : particles is (LS, BS, NHID, NPART); h = cols [0,MID), this particle = col p
  for (int i = tid; i < LS * BS * NHID; i += 256) {
    int n = i & 255, b = (i >> 8) & 15, l = i >> 12;
    float hv = particles[((l * BS + b) * NHID + n) * NPART + p];
    if (l == 0) h0bf[b][n] = f2bf(hv); else h1bf[b][n] = f2bf(hv);
  }

  // c state lives in registers as WMMA C-fragments (wave w owns unit tiles w and w+8)
  v8f c0f[2], c1f[2];
  #pragma unroll
  for (int up = 0; up < 2; ++up) {
    int col = 16 * (w + 8 * up) + nlo;
    #pragma unroll
    for (int r = 0; r < 8; ++r) {
      int m = r + 8 * g;
      c0f[up][r] = particles[((0 * BS + m) * NHID + col) * NPART + p + MID];
      c1f[up][r] = particles[((1 * BS + m) * NHID + col) * NPART + p + MID];
    }
  }
  float bc0[8], bc1[8];
  #pragma unroll
  for (int j = 0; j < 8; ++j) {
    int col = 16 * (w + 8 * j) + nlo;
    bc0[j] = b0[col];
    bc1[j] = b1[col];
  }
  const float qv = fabsf(qp[0]);
  const float evv = fabsf(ep[0]);
  const int bofs = (w * 32 + lane) * 16;   // per-wave/lane base into packed weights
  __syncthreads();

  #pragma clang loop unroll(disable)
  for (int t = 0; t < TSTEPS; ++t) {
    // Opaque zero derived from t: blocks LICM from hoisting the (t-invariant) weight
    // loads out of the time loop (which previously spilled ~1.7KB/lane to scratch),
    // while keeping pointer provenance so the loads stay global_load_b128.
    int opq = t;
    asm volatile("" : "+v"(opq));
    const int tofs = opq - t;   // == 0, but not provably so

    // ================= layer 0 : K = 64 (x_t) + 256 (h0) =================
    v8f acc[8];
    #pragma unroll
    for (int j = 0; j < 8; ++j) {
      v8f a;
      #pragma unroll
      for (int r = 0; r < 8; ++r) a[r] = bc0[j];
      acc[j] = a;
    }
    #pragma unroll
    for (int kk = 0; kk < 10; ++kk) {
      const unsigned short* arow = (kk < 2) ? &xbf[t][nlo][32 * kk + 8 * g]
                                            : &h0bf[nlo][32 * (kk - 2) + 8 * g];
      v16bf A = load_a_frag(arow);
      const unsigned short* bp = wpk0 + (kk * 32768 + bofs + tofs);
      v16bf Bf[8];
      #pragma unroll
      for (int j = 0; j < 8; ++j) Bf[j] = load_b_frag(bp + j * 4096);
      #pragma unroll
      for (int j = 0; j < 8; ++j)
        acc[j] = __builtin_amdgcn_wmma_f32_16x16x32_bf16(false, A, false, Bf[j], (short)0, acc[j], false, false);
    }
    // gates: j=0,1 -> i ; 2,3 -> f ; 4,5 -> g ; 6,7 -> o (unit tiles w, w+8)
    float h0v[2][8];
    #pragma unroll
    for (int up = 0; up < 2; ++up) {
      #pragma unroll
      for (int r = 0; r < 8; ++r) {
        float iv = sigm(acc[0 + up][r]);
        float fv = sigm(acc[2 + up][r]);
        float gv = tanh_f(acc[4 + up][r]);
        float ov = sigm(acc[6 + up][r]);
        float cv = fv * c0f[up][r] + iv * gv;
        c0f[up][r] = cv;
        h0v[up][r] = ov * tanh_f(cv);
      }
    }
    __syncthreads();  // all waves done reading h0bf
    #pragma unroll
    for (int up = 0; up < 2; ++up) {
      int col = 16 * (w + 8 * up) + nlo;
      #pragma unroll
      for (int r = 0; r < 8; ++r) h0bf[r + 8 * g][col] = f2bf(h0v[up][r]);
    }
    if (t == TSTEPS - 1) {
      #pragma unroll
      for (int up = 0; up < 2; ++up) {
        int col = 16 * (w + 8 * up) + nlo;
        #pragma unroll
        for (int r = 0; r < 8; ++r) {
          int m = r + 8 * g;
          pr[((0 * BS + m) * NHID + col) * NPART + p] =
              h0v[up][r] + qv * eps_h[((p * LS + 0) * BS + m) * NHID + col];
        }
      }
    }
    __syncthreads();

    // ================= layer 1 : K = 256 (h0_new) + 256 (h1) =================
    #pragma unroll
    for (int j = 0; j < 8; ++j) {
      v8f a;
      #pragma unroll
      for (int r = 0; r < 8; ++r) a[r] = bc1[j];
      acc[j] = a;
    }
    #pragma unroll
    for (int kk = 0; kk < 16; ++kk) {
      const unsigned short* arow = (kk < 8) ? &h0bf[nlo][32 * kk + 8 * g]
                                            : &h1bf[nlo][32 * (kk - 8) + 8 * g];
      v16bf A = load_a_frag(arow);
      const unsigned short* bp = wpk1 + (kk * 32768 + bofs + tofs);
      v16bf Bf[8];
      #pragma unroll
      for (int j = 0; j < 8; ++j) Bf[j] = load_b_frag(bp + j * 4096);
      #pragma unroll
      for (int j = 0; j < 8; ++j)
        acc[j] = __builtin_amdgcn_wmma_f32_16x16x32_bf16(false, A, false, Bf[j], (short)0, acc[j], false, false);
    }
    float h1v[2][8];
    #pragma unroll
    for (int up = 0; up < 2; ++up) {
      #pragma unroll
      for (int r = 0; r < 8; ++r) {
        float iv = sigm(acc[0 + up][r]);
        float fv = sigm(acc[2 + up][r]);
        float gv = tanh_f(acc[4 + up][r]);
        float ov = sigm(acc[6 + up][r]);
        float cv = fv * c1f[up][r] + iv * gv;
        c1f[up][r] = cv;
        h1v[up][r] = ov * tanh_f(cv);
      }
    }
    __syncthreads();
    #pragma unroll
    for (int up = 0; up < 2; ++up) {
      int col = 16 * (w + 8 * up) + nlo;
      #pragma unroll
      for (int r = 0; r < 8; ++r) h1bf[r + 8 * g][col] = f2bf(h1v[up][r]);
    }
    if (t == TSTEPS - 1) {
      #pragma unroll
      for (int up = 0; up < 2; ++up) {
        int col = 16 * (w + 8 * up) + nlo;
        #pragma unroll
        for (int r = 0; r < 8; ++r) {
          int m = r + 8 * g;
          pr[((1 * BS + m) * NHID + col) * NPART + p] =
              h1v[up][r] + qv * eps_h[((p * LS + 1) * BS + m) * NHID + col];
        }
      }
    }
    __syncthreads();
  }

  // final c: pr c-part (column p + MID), cT = c + |e| * eps_c
  #pragma unroll
  for (int up = 0; up < 2; ++up) {
    int col = 16 * (w + 8 * up) + nlo;
    #pragma unroll
    for (int r = 0; r < 8; ++r) {
      int m = r + 8 * g;
      pr[((0 * BS + m) * NHID + col) * NPART + p + MID] =
          c0f[up][r] + evv * eps_c[((p * LS + 0) * BS + m) * NHID + col];
      pr[((1 * BS + m) * NHID + col) * NPART + p + MID] =
          c1f[up][r] + evv * eps_c[((p * LS + 1) * BS + m) * NHID + col];
    }
  }
}

// ---------------- obs: Y[b,N] = sum_n Wh[n] * pr[1,b,n,N] + bh ----------------
__global__ void obs_kernel(const float* __restrict__ pr, const float* __restrict__ Wh,
                           const float* __restrict__ bh, float* __restrict__ Y) {
  int id = blockIdx.x * 256 + threadIdx.x;
  if (id >= BS * NPART) return;
  int b = id >> 9, N = id & 511;
  float acc = bh[0];
  const float* base = pr + ((LS - 1) * BS + b) * NHID * NPART + N;
  for (int n = 0; n < NHID; ++n) acc += Wh[n] * base[n * NPART];
  Y[id] = acc;
}

// -------- block reduction helpers --------
__device__ float block_sum(float v, float* red) {
  int tid = threadIdx.x;
  __syncthreads();
  red[tid] = v;
  __syncthreads();
  for (int s = 128; s > 0; s >>= 1) {
    if (tid < s) red[tid] += red[tid + s];
    __syncthreads();
  }
  return red[0];
}
__device__ float block_max(float v, float* red) {
  int tid = threadIdx.x;
  __syncthreads();
  red[tid] = v;
  __syncthreads();
  for (int s = 128; s > 0; s >>= 1) {
    if (tid < s) red[tid] = fmaxf(red[tid], red[tid + s]);
    __syncthreads();
  }
  return red[0];
}

// ---------------- per-batch stats + unnormalized log-weights (M=1 scalar path) ----------------
__global__ __launch_bounds__(256) void stats1_kernel(const float* __restrict__ Y,
                                                     const float* __restrict__ weights,
                                                     const float* __restrict__ y,
                                                     const float* __restrict__ rp,
                                                     float* __restrict__ unnw) {
  __shared__ float red[256];
  int b = blockIdx.x, tid = threadIdx.x;
  float wl0 = logf(weights[b * NPART + tid]);
  float wl1 = logf(weights[b * NPART + tid + 256]);
  float Y0 = Y[b * NPART + tid], Y1 = Y[b * NPART + tid + 256];
  float wsum = block_sum(wl0 + wl1, red);
  float syw = block_sum(wl0 * Y0 + wl1 * Y1, red);
  float Wmu = syw / wsum;
  float c0 = Y0 - Wmu, c1 = Y1 - Wmu;
  float scov = block_sum(c0 * c0 + c1 * c1, red);
  float r = rp[0];
  float sigp = scov / (float)(NPART - 1) + r * r + 1e-6f;  // sig_e + 1e-6 (inv_logdet)
  float inv = 1.0f / sigp;
  float ld = logf(sigp);
  float yb = y[b];
  float i0 = yb - Y0 * wl0 / wsum;
  float i1 = yb - Y1 * wl1 / wsum;
  unnw[b * NPART + tid]       = wl0 - 0.5f * ld - 0.5f * (i0 * i0 * inv);
  unnw[b * NPART + tid + 256] = wl1 - 0.5f * ld - 0.5f * (i1 * i1 * inv);
}

// ---------------- resample: one block per (b, half); bitonic argsort of 256 keys ----------------
__global__ __launch_bounds__(256) void resample_kernel(const float* __restrict__ unnw,
                                                       const float* __restrict__ gumbel,
                                                       float* __restrict__ wf_out,
                                                       int* __restrict__ idxtab,
                                                       int* __restrict__ doflag) {
  __shared__ float skey[256];
  __shared__ int sidx[256];
  __shared__ float red[256];
  int b = blockIdx.x >> 1, half = blockIdx.x & 1, j = threadIdx.x;
  int col = half * 256 + j;

  // normalize across batch axis: new_w[b,N] = exp(unnw[b,N] - log(sum_b exp(unnw[b,N])))
  float S = 0.0f;
  for (int bb = 0; bb < BS; ++bb) S += expf(unnw[bb * NPART + col]);
  float wh = expf(unnw[b * NPART + col] - logf(S));

  float s2 = block_sum(wh * wh, red);
  int do_rs = ((1.0f / s2) < (float)NPART / 4.0f) ? 1 : 0;

  float soft = 0.5f * wh + 0.5f / (float)MID;
  float key = logf(soft) + gumbel[(b * 2 + half) * MID + j];
  skey[j] = key;
  sidx[j] = j;
  __syncthreads();
  // bitonic sort, descending by key (argsort(-(key)))
  for (int k = 2; k <= 256; k <<= 1) {
    for (int jj = k >> 1; jj > 0; jj >>= 1) {
      int ixj = j ^ jj;
      if (ixj > j) {
        bool desc = ((j & k) == 0);
        float a = skey[j], c = skey[ixj];
        bool sw = desc ? (a < c) : (a > c);
        if (sw) {
          skey[j] = c; skey[ixj] = a;
          int tix = sidx[j]; sidx[j] = sidx[ixj]; sidx[ixj] = tix;
        }
      }
      __syncthreads();
    }
  }
  idxtab[(b * 2 + half) * MID + j] = half * 256 + sidx[j];  // global source column
  if (j == 0) doflag[b * 2 + half] = do_rs;

  // w_r = softmax(log(wh/soft))  (unpermuted, as in reference)
  float lw = logf(wh / soft);
  float mx = block_max(lw, red);
  float se = block_sum(expf(lw - mx), red);
  float w_r = expf(lw - mx - logf(se));
  wf_out[b * NPART + col] = do_rs ? w_r : wh;
}

// ---------------- gather pf[l,b,n,col] = pr[l,b,n, src] ----------------
__global__ void gather_kernel(const float* __restrict__ pr, const int* __restrict__ idxtab,
                              const int* __restrict__ doflag, float* __restrict__ pf) {
  int id = blockIdx.x * 256 + threadIdx.x;
  if (id >= LS * BS * NHID * NPART) return;
  int col = id & 511;
  int row = id >> 9;             // (l*BS + b)*NHID + n
  int b = (row >> 8) & 15;
  int half = col >> 8, j = col & 255;
  int src = doflag[b * 2 + half] ? idxtab[(b * 2 + half) * MID + j] : col;
  pf[id] = pr[row * NPART + src];
}

// ---------------- final estimate + likelihood (Yf = gather of Y; M=1 scalar path) ----------------
__global__ __launch_bounds__(256) void final_kernel(const float* __restrict__ Y,
                                                    const float* __restrict__ y,
                                                    const float* __restrict__ rp,
                                                    const int* __restrict__ idxtab,
                                                    const int* __restrict__ doflag,
                                                    const float* __restrict__ wf,
                                                    float* __restrict__ out_Wmu,
                                                    float* __restrict__ out_sig,
                                                    float* __restrict__ out_l,
                                                    float* __restrict__ out_nis) {
  __shared__ float red[256];
  int tid = threadIdx.x;
  float l_acc = 0.0f, nis_acc = 0.0f;
  float r = rp[0];
  for (int b = 0; b < BS; ++b) {
    float vals[2], wfv[2];
    #pragma unroll
    for (int h = 0; h < 2; ++h) {
      int col = tid + 256 * h;
      int half = col >> 8, j = col & 255;
      int src = doflag[b * 2 + half] ? idxtab[(b * 2 + half) * MID + j] : col;
      vals[h] = Y[b * NPART + src];
      wfv[h] = wf[b * NPART + col];
    }
    float sw = block_sum(wfv[0] + wfv[1], red);
    float syw = block_sum(vals[0] * wfv[0] + vals[1] * wfv[1], red);
    float Wmu = syw / sw;
    float c0 = vals[0] - Wmu, c1 = vals[1] - Wmu;
    float sc = block_sum(c0 * c0 + c1 * c1, red);
    float sig = sc / (float)(NPART - 1) + r * r;
    float sigp = sig + 1e-6f;
    float inv = 1.0f / sigp, ld = logf(sigp);
    float inn = y[b] - Wmu;
    float qf = inn * inn * inv;
    if (tid == 0) { out_Wmu[b] = Wmu; out_sig[b] = sig; }
    l_acc += -0.5f * ld - 0.5f * qf;
    nis_acc += qf;
  }
  if (tid == 0) { out_l[0] = l_acc; out_nis[0] = nis_acc / (float)BS; }
}

extern "C" void kernel_launch(void* const* d_in, const int* in_sizes, int n_in,
                              void* d_out, int out_size, void* d_ws, size_t ws_size,
                              hipStream_t stream) {
  (void)in_sizes; (void)n_in; (void)out_size; (void)ws_size;
  const float* x        = (const float*)d_in[0];
  const float* y        = (const float*)d_in[1];
  const float* particles= (const float*)d_in[2];
  const float* weights  = (const float*)d_in[3];
  const float* q        = (const float*)d_in[4];
  const float* e        = (const float*)d_in[5];
  const float* r        = (const float*)d_in[6];
  const float* Wih0     = (const float*)d_in[7];
  const float* Whh0     = (const float*)d_in[8];
  const float* b0       = (const float*)d_in[9];
  const float* Wih1     = (const float*)d_in[10];
  const float* Whh1     = (const float*)d_in[11];
  const float* b1       = (const float*)d_in[12];
  const float* Wh       = (const float*)d_in[13];
  const float* bh       = (const float*)d_in[14];
  const float* eps_h    = (const float*)d_in[15];
  const float* eps_c    = (const float*)d_in[16];
  const float* gumbel   = (const float*)d_in[17];

  // workspace layout (bytes)
  char* ws = (char*)d_ws;
  unsigned short* wpk0 = (unsigned short*)(ws + 0);          // 655360 B
  unsigned short* wpk1 = (unsigned short*)(ws + 655360);     // 1048576 B
  float* pr   = (float*)(ws + 1703936);                      // 16777216 B
  float* Y    = (float*)(ws + 18481152);                     // 32768 B
  float* unnw = (float*)(ws + 18513920);                     // 32768 B
  int*   idxt = (int*)  (ws + 18546688);                     // 32768 B
  int*   dofl = (int*)  (ws + 18579456);                     // 128 B

  float* out = (float*)d_out;
  float* out_Wmu = out;                  // 16
  float* out_sig = out + 16;             // 16
  float* out_pf  = out + 32;             // 4194304
  float* out_wf  = out + 32 + LS * BS * NHID * NPART;   // 8192
  float* out_l   = out_wf + BS * NPART;  // 1
  float* out_nis = out_l + 1;            // 1

  pack_weights_kernel<<<3328, 256, 0, stream>>>(Wih0, Whh0, Wih1, Whh1, wpk0, wpk1);
  lstm_particle_kernel<<<MID, 256, 0, stream>>>(x, particles, q, e, wpk0, b0, wpk1, b1,
                                                eps_h, eps_c, pr);
  obs_kernel<<<(BS * NPART) / 256, 256, 0, stream>>>(pr, Wh, bh, Y);
  stats1_kernel<<<BS, 256, 0, stream>>>(Y, weights, y, r, unnw);
  resample_kernel<<<BS * 2, 256, 0, stream>>>(unnw, gumbel, out_wf, idxt, dofl);
  gather_kernel<<<(LS * BS * NHID * NPART) / 256, 256, 0, stream>>>(pr, idxt, dofl, out_pf);
  final_kernel<<<1, 256, 0, stream>>>(Y, y, r, idxt, dofl, out_wf, out_Wmu, out_sig, out_l, out_nis);
}
